// Sparse3DNA_22144851378712
// MI455X (gfx1250) — matrix operand order repaired and checked by reading, not compiled, not run on hardware
//
#include <hip/hip_runtime.h>
#include <hip/hip_bf16.h>

typedef __attribute__((ext_vector_type(16))) _Float16 v16h;
typedef __attribute__((ext_vector_type(8)))  float    v8f;
typedef __attribute__((__vector_size__(16))) int      i4vec;

#define DIMM   512
#define NSEQ   4097      // padded tokens per batch
#define TTOK   4096
#define BATCH  2
#define NHEADS 8
#define DHEAD  64

#define AS1 __attribute__((address_space(1)))
#define AS3 __attribute__((address_space(3)))

#if __has_builtin(__builtin_amdgcn_global_load_async_to_lds_b128)
#define HAVE_ASYNC_LDS 1
#else
#define HAVE_ASYNC_LDS 0
#endif

__device__ __forceinline__ void wait_asynccnt0() {
#if __has_builtin(__builtin_amdgcn_s_wait_asynccnt)
    __builtin_amdgcn_s_wait_asynccnt(0);
#else
    asm volatile("s_wait_asynccnt 0x0" ::: "memory");
#endif
}

// ---------------------------------------------------------------------------
// WMMA GEMM: C[M,N] = A[M,Kd] * B[Kd,N] (+ bias), A/B f16 row-major, C f32.
// Block = 256 threads = 8 waves; block tile 64x64; wave (wy 0..3, wx 0..1)
// owns a 16x32 strip -> 2 x v_wmma_f32_16x16x32_f16 per K-step, A-frag reused.
// A tile staged via GLOBAL_LOAD_ASYNC_TO_LDS_B128 (ASYNCcnt) when available.
// ---------------------------------------------------------------------------
#define BM 64
#define BN 64
#define BK 32
#define LDP 40   // padded LDS row stride (halves); 80B = multiple of 16B

__global__ __launch_bounds__(256) void gemm_wmma_f16(
    const _Float16* __restrict__ A, const _Float16* __restrict__ Bw,
    float* __restrict__ C, int M, int N, int Kd, const float* __restrict__ bias)
{
    __shared__ _Float16 sA[BM][LDP];   // M x K
    __shared__ _Float16 sB[BN][LDP];   // N x K (transposed stage)

    const int tid  = threadIdx.x;
    const int wv   = tid >> 5;
    const int lane = tid & 31;
    const int wy   = wv >> 1;     // 0..3 : M sub-tile (16 rows)
    const int wx   = wv & 1;      // 0..1 : N strip (32 cols)
    const int lh   = lane >> 4;   // lane half
    const int lr   = lane & 15;

    const int blockM = blockIdx.x * BM;
    const int blockN = blockIdx.y * BN;

    // per-thread staging coordinates
    const int ar = tid >> 2;            // A row in tile (0..63)
    const int ac = (tid & 3) * 8;       // A col (0,8,16,24)
    const int gr = blockM + ar;
    const int bk = tid >> 3;            // B k row in tile (0..31)
    const int bn = (tid & 7) * 8;       // B col (0..56 step 8)

    const _Float16* aSrc = A  + (size_t)gr * Kd + ac;            // valid iff gr<M
    const _Float16* bSrc = Bw + (size_t)bk * N + (blockN + bn);

    v8f acc0 = {};
    v8f acc1 = {};

    for (int k0 = 0; k0 < Kd; k0 += BK) {
        // ---- stage A tile (64x32, untransposed): 16B per thread ----
        if (gr < M) {
#if HAVE_ASYNC_LDS
            __builtin_amdgcn_global_load_async_to_lds_b128(
                (AS1 i4vec*)(aSrc + k0), (AS3 i4vec*)&sA[ar][ac], 0, 0);
#else
            *(uint4*)&sA[ar][ac] = *(const uint4*)(aSrc + k0);
#endif
        } else {
            uint4 z = {0, 0, 0, 0};
            *(uint4*)&sA[ar][ac] = z;
        }
        // ---- stage B tile (32x64) transposed into sB[n][k] ----
        {
            uint4 braw = *(const uint4*)(bSrc + (size_t)k0 * N);
            const _Float16* bh = (const _Float16*)&braw;
            #pragma unroll
            for (int e = 0; e < 8; ++e)
                sB[bn + e][bk] = bh[e];
        }
        if (k0 + BK < Kd) {   // prefetch next K-tiles -> global_prefetch_b8
            if (gr < M) __builtin_prefetch(aSrc + k0 + BK, 0, 1);
            __builtin_prefetch(bSrc + (size_t)(k0 + BK) * N, 0, 1);
        }
#if HAVE_ASYNC_LDS
        wait_asynccnt0();
#endif
        __syncthreads();

        // ---- build fragments (contiguous k = 16*lh + e for both A and B) ----
        v16h af, bf0, bf1;
        const _Float16* aB  = &sA[wy * 16 + lr][0];
        #pragma unroll
        for (int e = 0; e < 16; ++e) {
            int v = e >> 1, w = e & 1;
            int kk = lh * 16 + 2 * v + w;
            af[e] = aB[kk];
        }
        const _Float16* bB0 = &sB[wx * 32 + lr][0];
        const _Float16* bB1 = &sB[wx * 32 + 16 + lr][0];
        #pragma unroll
        for (int e = 0; e < 16; ++e) {
            bf0[e] = bB0[lh * 16 + e];
            bf1[e] = bB1[lh * 16 + e];
        }

        acc0 = __builtin_amdgcn_wmma_f32_16x16x32_f16(
                   false, af, false, bf0, (short)0, acc0, false, false);
        acc1 = __builtin_amdgcn_wmma_f32_16x16x32_f16(
                   false, af, false, bf1, (short)0, acc1, false, false);
        __syncthreads();
    }

    const int n0 = blockN + wx * 32 + lr;
    const int n1 = n0 + 16;
    const float b0 = bias ? bias[n0] : 0.0f;
    const float b1 = bias ? bias[n1] : 0.0f;
    #pragma unroll
    for (int r = 0; r < 8; ++r) {
        int m = blockM + wy * 16 + lh * 8 + r;
        if (m < M) {
            C[(size_t)m * N + n0] = acc0[r] + b0;
            C[(size_t)m * N + n1] = acc1[r] + b1;
        }
    }
}

// ---------------------------------------------------------------------------
// Pad x (B x 4096 x 512) to Xp f16 (B*4097 x 512), zero pad-row per batch.
// ---------------------------------------------------------------------------
__global__ __launch_bounds__(256) void build_xp(
    const float* __restrict__ x, _Float16* __restrict__ Xp)
{
    int idx = blockIdx.x * blockDim.x + threadIdx.x;   // over 8194*512
    int row = idx >> 9;
    int c   = idx & 511;
    int b   = row / NSEQ;
    int t   = row % NSEQ;
    float v = (t < TTOK) ? x[((size_t)b * TTOK + t) * DIMM + c] : 0.0f;
    Xp[idx] = (_Float16)v;
}

__global__ __launch_bounds__(256) void cvt_f32_f16(
    const float* __restrict__ src, _Float16* __restrict__ dst, int n)
{
    int i = blockIdx.x * blockDim.x + threadIdx.x;
    if (i < n) dst[i] = (_Float16)src[i];
}

// ---------------------------------------------------------------------------
// Windowed causal attention, one wave32 per query.
// Q  : f32 (B*NSEQ x 512), row (b*NSEQ + i + 1) is query i (already projected)
// KV : f32 (B*NSEQ x 1024), cols [0,512) = K heads, [512,1024) = V heads
// O  : f16 (B*TTOK x 512), row (b*TTOK + i + 1) gets query i's output
// ---------------------------------------------------------------------------
__global__ __launch_bounds__(256) void attn_win_kernel(
    const float* __restrict__ Q, const float* __restrict__ KV,
    _Float16* __restrict__ O)
{
    const int wv   = threadIdx.x >> 5;
    const int lane = threadIdx.x & 31;
    const int gid  = blockIdx.x * 8 + wv;          // 0..65519
    const int i    = gid % 4095;                   // queries 0..4094 kept
    const int h    = (gid / 4095) % NHEADS;
    const int b    = gid / (4095 * NHEADS);

    // lane j owns key slot j (j==0: BOS, 1..27: 3x3x3 window)
    int  tok    = b * NSEQ;      // safe default = BOS row
    bool masked = false;
    if (lane == 0) {
        // BOS key: token 0, never masked
    } else if (lane < 28) {
        int jj = lane - 1;
        int df = jj / 9 - 1, dy = (jj / 3) % 3 - 1, dx = jj % 3 - 1;
        int f  = i >> 10, y = (i >> 5) & 31, xx = i & 31;
        int nf = f + df, ny = y + dy, nx = xx + dx;
        if (nf < 0 || nf >= 4 || ny < 0 || ny >= 32 || nx < 0 || nx >= 32) {
            masked = true;                          // WIN_IDX == T
        } else {
            int w  = ((nf << 5) + ny) * 32 + nx;
            masked = (i < w);                       // causal
            if (!masked) tok = b * NSEQ + w + 1;    // shifted token index
        }
    } else {
        masked = true;                              // lanes 28..31 inert
    }

    const float* qp = Q  + ((size_t)(b * NSEQ + i + 1)) * DIMM + h * DHEAD;
    const float* kp = KV + (size_t)tok * (2 * DIMM) + h * DHEAD;

    float s;
    if (!masked) {
        float acc = 0.0f;
        #pragma unroll
        for (int d = 0; d < DHEAD; d += 4) {
            float4 qv = *(const float4*)(qp + d);
            float4 kv = *(const float4*)(kp + d);
            acc += qv.x * kv.x + qv.y * kv.y + qv.z * kv.z + qv.w * kv.w;
        }
        s = acc * 0.125f;                           // DIM_HEAD^-0.5
    } else {
        s = -3.402823466e38f;
    }

    // wave softmax over 32 lanes (inert lanes contribute exp(-max)=0)
    float m = s;
    #pragma unroll
    for (int o = 16; o > 0; o >>= 1) m = fmaxf(m, __shfl_xor(m, o, 32));
    float p = __expf(s - m);
    float sum = p;
    #pragma unroll
    for (int o = 16; o > 0; o >>= 1) sum += __shfl_xor(sum, o, 32);
    const float attn = p / sum;

    // out[d] = sum_j attn_j * V_j[d]; each lane owns dims (lane, lane+32)
    const float* Vbase = KV + DIMM;
    float o0 = 0.0f, o1 = 0.0f;
    #pragma unroll 4
    for (int j = 0; j < 28; ++j) {
        float a  = __shfl(attn, j, 32);
        int   tj = __shfl(tok,  j, 32);
        const float* vp = Vbase + (size_t)tj * (2 * DIMM) + h * DHEAD;
        o0 += a * vp[lane];
        o1 += a * vp[lane + 32];
    }

    size_t orow = ((size_t)b * TTOK + (i + 1)) * DIMM + h * DHEAD;
    O[orow + lane]      = (_Float16)o0;
    O[orow + lane + 32] = (_Float16)o1;
}

// O row 0 per batch = bos_v = V(token 0) across all heads
__global__ __launch_bounds__(256) void bos_copy(
    const float* __restrict__ KV, _Float16* __restrict__ O)
{
    int c = blockIdx.x * blockDim.x + threadIdx.x;
    if (c >= BATCH * DIMM) return;
    int b = c / DIMM, d = c % DIMM;
    O[(size_t)b * TTOK * DIMM + d] =
        (_Float16)KV[((size_t)b * NSEQ) * (2 * DIMM) + DIMM + d];
}

// ---------------------------------------------------------------------------
extern "C" void kernel_launch(void* const* d_in, const int* in_sizes, int n_in,
                              void* d_out, int out_size, void* d_ws, size_t ws_size,
                              hipStream_t stream)
{
    const float* x   = (const float*)d_in[0];   // (2, 4096, 512)
    const float* Wq  = (const float*)d_in[1];   // (512, 512)
    const float* Wkv = (const float*)d_in[2];   // (512, 1024)
    const float* Wo  = (const float*)d_in[3];   // (512, 512)
    const float* bo  = (const float*)d_in[4];   // (512,)
    float* out = (float*)d_out;                 // (2, 4096, 512)

    const int Mpad = BATCH * NSEQ;              // 8194
    const int Mout = BATCH * TTOK;              // 8192

    char* ws = (char*)d_ws;
    size_t off = 0;
    auto take = [&](size_t bytes) -> char* {
        char* p = ws + off;
        off = (off + bytes + 255) & ~(size_t)255;
        return p;
    };
    _Float16* Xp    = (_Float16*)take((size_t)Mpad * DIMM * 2);       // 8.4 MB
    _Float16* Wq16  = (_Float16*)take((size_t)DIMM * DIMM * 2);       // 0.5 MB
    _Float16* Wkv16 = (_Float16*)take((size_t)DIMM * 2 * DIMM * 2);   // 1.0 MB
    _Float16* Wo16  = (_Float16*)take((size_t)DIMM * DIMM * 2);       // 0.5 MB
    float*    Qbuf  = (float*)take((size_t)Mpad * DIMM * 4);          // 16.8 MB
    float*    KVbuf = (float*)take((size_t)Mpad * 2 * DIMM * 4);      // 33.6 MB
    _Float16* Obuf  = (_Float16*)take((size_t)Mout * DIMM * 2);       // 8.4 MB

    // 1) pad + convert inputs/weights to f16
    {
        int n = Mpad * DIMM;                    // 4,195,328 (multiple of 256)
        build_xp<<<n / 256, 256, 0, stream>>>(x, Xp);
    }
    cvt_f32_f16<<<(DIMM * DIMM + 255) / 256, 256, 0, stream>>>(Wq, Wq16, DIMM * DIMM);
    cvt_f32_f16<<<(DIMM * 2 * DIMM + 255) / 256, 256, 0, stream>>>(Wkv, Wkv16, DIMM * 2 * DIMM);
    cvt_f32_f16<<<(DIMM * DIMM + 255) / 256, 256, 0, stream>>>(Wo, Wo16, DIMM * DIMM);

    // 2) Q = Xp @ Wq ; KV = Xp @ Wkv   (WMMA)
    {
        dim3 gq((Mpad + BM - 1) / BM, DIMM / BN);        // (129, 8)
        gemm_wmma_f16<<<gq, 256, 0, stream>>>(Xp, Wq16, Qbuf, Mpad, DIMM, DIMM, nullptr);
        dim3 gkv((Mpad + BM - 1) / BM, (2 * DIMM) / BN); // (129, 16)
        gemm_wmma_f16<<<gkv, 256, 0, stream>>>(Xp, Wkv16, KVbuf, Mpad, 2 * DIMM, DIMM, nullptr);
    }

    // 3) windowed attention (queries 0..4094 per (b,h)) + BOS rows
    bos_copy<<<(BATCH * DIMM + 255) / 256, 256, 0, stream>>>(KVbuf, Obuf);
    {
        int nq = BATCH * NHEADS * 4095;         // 65520 = 8 waves * 8190 blocks
        attn_win_kernel<<<nq / 8, 256, 0, stream>>>(Qbuf, KVbuf, Obuf);
    }

    // 4) out = O @ Wo + bo  (WMMA, writes d_out directly)
    {
        dim3 go(Mout / BM, DIMM / BN);          // (128, 8)
        gemm_wmma_f16<<<go, 256, 0, stream>>>(Obuf, Wo16, out, Mout, DIMM, DIMM, bo);
    }
}
